// VisionTransformer_996432412678
// MI455X (gfx1250) — compile-verified, matrix operand order checked by reading, not run on hardware
//
#include <hip/hip_runtime.h>
#include <hip/hip_bf16.h>
#include <math.h>

// ---------------------------------------------------------------------------
// ViT-B/16 forward for MI455X (gfx1250): all GEMMs via v_wmma_f32_16x16x32_bf16,
// activation tiles staged with global_load_async_to_lds_b128 (ASYNCcnt path).
// ---------------------------------------------------------------------------

#define B_SZ   32
#define GRID_P 14
#define NTOK   197          // 196 patches + cls
#define DMODEL 768
#define HEADS  12
#define DH     64
#define FF_DIM 3072
#define DEPTH  12
#define MROWS  (B_SZ * NTOK)   // 6304

typedef __bf16 v16bf __attribute__((ext_vector_type(16)));
typedef __bf16 v8bf  __attribute__((ext_vector_type(8)));
typedef float  v8f   __attribute__((ext_vector_type(8)));

__device__ __forceinline__ v8f wmma_bf16(v16bf a, v16bf b, v8f c) {
    // D = A(16x32 bf16) * B(32x16 bf16) + C(16x16 f32)
    return __builtin_amdgcn_wmma_f32_16x16x32_bf16(
        /*neg_a=*/false, a, /*neg_b=*/false, b,
        /*c_mod=*/(short)0, c, /*reuse_a=*/false, /*reuse_b=*/false);
}

// CDNA5 16-bit A-fragment layout (ISA 7.12.2): lane L in [0,16): row M=L,
// holds K = {0..7, 16..23}; lane L+16: row M=L, K = {8..15, 24..31}.
// B-fragment for C = A * W^T with row-major W[n][k] uses the identical
// pattern with "row" = output column n.
__device__ __forceinline__ v16bf frag_ld(const __bf16* __restrict__ p,
                                         int stride, int lane) {
    const int r  = lane & 15;
    const int kh = (lane >> 4) & 1;
    const __bf16* q = p + r * stride + kh * 8;
    v8bf lo = *(const v8bf*)(q);
    v8bf hi = *(const v8bf*)(q + 16);
    return __builtin_shufflevector(lo, hi, 0,1,2,3,4,5,6,7,
                                           8,9,10,11,12,13,14,15);
}

// ---- CDNA5 async memory -> LDS copy (ISA 10.x / CDNA5 ISA) --------
// GV mode: per-lane 16B copy, LDS dest offset from VDST VGPR, 64-bit global
// address from VGPR pair. Tracked on ASYNCcnt.
__device__ __forceinline__ void async_ld_b128(void* lds, const void* gptr) {
    const unsigned      ldso = (unsigned)(uintptr_t)lds;       // flat->LDS: low 32 bits
    const unsigned long long ga = (unsigned long long)(uintptr_t)gptr;
    asm volatile("global_load_async_to_lds_b128 %0, %1, off"
                 :: "v"(ldso), "v"(ga) : "memory");
}
__device__ __forceinline__ void wait_async0() {
    asm volatile("s_wait_asynccnt 0x0" ::: "memory");
}

// ---------------------------------------------------------------------------
// Generic GEMM: C[M,N] = A[M,K](bf16) * W[N,K](f32->bf16)^T + bias
//   flags bit0: exact-erf GELU epilogue
//   flags bit1: ViT row remap for out32 (skip cls rows): orow = r + r/196 + 1
//   res (optional, fp32): residual added (indexed with orow)
//   out32 (optional, fp32) / out16 (optional, bf16, indexed with r)
// ---------------------------------------------------------------------------
#define BM 128
#define BN 128
#define BK 32
#define LDT 40   // padded LDS row stride (bf16 elems)

__global__ __launch_bounds__(256)
void k_gemm(const __bf16* __restrict__ A, const float* __restrict__ W,
            const float* __restrict__ bias, const float* __restrict__ res,
            float* __restrict__ out32, __bf16* __restrict__ out16,
            int M, int N, int K, int flags)
{
    __shared__ __bf16 sA[BM][LDT];
    __shared__ __bf16 sB[BN][LDT];

    const int t    = threadIdx.x;
    const int lane = t & 31;
    const int w    = t >> 5;
    const int wm   = w & 3;        // 4 waves along M (4*32 = 128)
    const int wn   = w >> 2;       // 2 waves along N (2*64 = 128)
    const int m0   = blockIdx.x * BM;
    const int n0   = blockIdx.y * BN;

    const int lrow  = t >> 1;        // 0..127
    const int lhalf = (t & 1) * 16;  // 0 or 16

    const bool fullM = (m0 + BM) <= M;   // uniform per block

    v8f acc[2][4] = {};

    for (int k0 = 0; k0 < K; k0 += BK) {
        // ---- stage A tile (bf16): async DMA to LDS for full blocks ----
        if (fullM) {
            const __bf16* ap = A + (size_t)(m0 + lrow) * K + k0 + lhalf;
            async_ld_b128(&sA[lrow][lhalf],     ap);
            async_ld_b128(&sA[lrow][lhalf + 8], ap + 8);
        } else {
            const int gr = m0 + lrow;
            v8bf a0 = {}, a1 = {};
            if (gr < M) {
                const __bf16* ap = A + (size_t)gr * K + k0 + lhalf;
                a0 = *(const v8bf*)(ap);
                a1 = *(const v8bf*)(ap + 8);
            }
            *(v8bf*)&sA[lrow][lhalf]     = a0;
            *(v8bf*)&sA[lrow][lhalf + 8] = a1;
        }
        // ---- prefetch next W tile (global_prefetch_b8) ----
        if (k0 + BK < K)
            __builtin_prefetch(W + (size_t)(n0 + lrow) * K + k0 + BK + lhalf, 0, 0);
        // ---- stage W tile (fp32 -> bf16) ----
        {
            const int gr = n0 + lrow;
            __bf16 tmp[16];
            if (gr < N) {
                const float* wp = W + (size_t)gr * K + k0 + lhalf;
                #pragma unroll
                for (int c = 0; c < 4; ++c) {
                    float4 f = *(const float4*)(wp + 4 * c);
                    tmp[4*c+0] = (__bf16)f.x; tmp[4*c+1] = (__bf16)f.y;
                    tmp[4*c+2] = (__bf16)f.z; tmp[4*c+3] = (__bf16)f.w;
                }
            } else {
                #pragma unroll
                for (int c = 0; c < 16; ++c) tmp[c] = (__bf16)0.f;
            }
            #pragma unroll
            for (int c = 0; c < 16; ++c) sB[lrow][lhalf + c] = tmp[c];
        }
        if (fullM) wait_async0();        // this wave's async LDS writes done
        __syncthreads();                 // all waves' tiles visible

        // ---- compute: 2x4 WMMA tiles per wave ----
        v16bf af[2], bf[4];
        #pragma unroll
        for (int i = 0; i < 2; ++i)
            af[i] = frag_ld(&sA[wm * 32 + i * 16][0], LDT, lane);
        #pragma unroll
        for (int j = 0; j < 4; ++j)
            bf[j] = frag_ld(&sB[wn * 64 + j * 16][0], LDT, lane);
        #pragma unroll
        for (int i = 0; i < 2; ++i)
            #pragma unroll
            for (int j = 0; j < 4; ++j)
                acc[i][j] = wmma_bf16(af[i], bf[j], acc[i][j]);
        __syncthreads();
    }

    // ---- epilogue (C layout: lane = col, vgpr v = row (+8 for hi half)) ----
    const bool do_gelu = (flags & 1) != 0;
    const bool remap   = (flags & 2) != 0;
    const int  crow    = (lane >> 4) << 3;  // 0 or 8
    const int  ccol    = lane & 15;

    #pragma unroll
    for (int i = 0; i < 2; ++i) {
        #pragma unroll
        for (int j = 0; j < 4; ++j) {
            const int gc = n0 + wn * 64 + j * 16 + ccol;
            #pragma unroll
            for (int v = 0; v < 8; ++v) {
                const int r = m0 + wm * 32 + i * 16 + crow + v;
                if (r < M && gc < N) {
                    float val = acc[i][j][v];
                    if (bias) val += bias[gc];
                    if (do_gelu)
                        val = 0.5f * val * (1.f + erff(val * 0.70710678118654752f));
                    const int orow = remap ? (r + r / 196 + 1) : r;
                    if (res)   val += res[(size_t)orow * N + gc];
                    if (out32) out32[(size_t)orow * N + gc] = val;
                    if (out16) out16[(size_t)r * N + gc] = (__bf16)val;
                }
            }
        }
    }
}

// ---------------------------------------------------------------------------
// Attention: one block per (batch, head); 4 waves, 16 query rows per wave.
// Scores kept in registers (13 v8f accs), softmax via cross-lane shuffles,
// probabilities staged bf16 in LDS, P*V via WMMA against LDS-transposed V.
// ---------------------------------------------------------------------------
#define KT   13    // ceil(197/16)
#define KP2  224   // keys padded to multiple of 32 for the PV K-loop
#define SVL  232   // padded LDS row stride

__global__ __launch_bounds__(128)
void k_attn(const __bf16* __restrict__ qkv, __bf16* __restrict__ o, float scale)
{
    __shared__ __bf16 sVT[DH][SVL];       // V transposed: [d][key]
    __shared__ __bf16 sP[4][16][SVL];     // per-wave probabilities

    const int bh   = blockIdx.x;
    const int b    = bh / HEADS;
    const int hd   = bh % HEADS;
    const int t    = threadIdx.x;
    const int lane = t & 31;
    const int w    = t >> 5;
    const size_t qkbase = (size_t)b * NTOK * (3 * DMODEL) + hd * DH;

    // stage V^T (zero-pad keys >= 197)
    for (int i = t; i < DH * KP2; i += 128) {
        const int d = i / KP2, n = i % KP2;
        __bf16 val = (__bf16)0.f;
        if (n < NTOK)
            val = qkv[qkbase + (size_t)n * (3 * DMODEL) + 2 * DMODEL + d];
        sVT[d][n] = val;
    }
    // zero the tail of this wave's P rows (cols 208..223 never written below)
    for (int i = lane; i < 16 * 16; i += 32)
        sP[w][i >> 4][208 + (i & 15)] = (__bf16)0.f;
    __syncthreads();

    const int r    = lane & 15;
    const int kh   = (lane >> 4) & 1;
    const int half = lane >> 4;

    for (int qt = w; qt < KT; qt += 4) {
        const int n0 = qt * 16;

        // ---- Q fragments (K = 64 -> two 16x32 chunks), zero-pad rows >=197
        v16bf qa[2];
        {
            const int qrow = n0 + r;
            #pragma unroll
            for (int f = 0; f < 2; ++f) {
                v8bf lo = {}, hi = {};
                if (qrow < NTOK) {
                    const __bf16* p = qkv + qkbase +
                        (size_t)qrow * (3 * DMODEL) + f * 32 + kh * 8;
                    lo = *(const v8bf*)(p);
                    hi = *(const v8bf*)(p + 16);
                }
                qa[f] = __builtin_shufflevector(lo, hi, 0,1,2,3,4,5,6,7,
                                                        8,9,10,11,12,13,14,15);
            }
        }

        // ---- scores: S[16 x 13*16] in registers ----
        v8f sc[KT];
        #pragma unroll
        for (int kt = 0; kt < KT; ++kt) {
            v8f s = {};
            const int krow = kt * 16 + r;           // key index for B-frag
            #pragma unroll
            for (int f = 0; f < 2; ++f) {
                v8bf lo = {}, hi = {};
                if (krow < NTOK) {
                    const __bf16* p = qkv + qkbase +
                        (size_t)krow * (3 * DMODEL) + DMODEL + f * 32 + kh * 8;
                    lo = *(const v8bf*)(p);
                    hi = *(const v8bf*)(p + 16);
                }
                v16bf kb = __builtin_shufflevector(lo, hi, 0,1,2,3,4,5,6,7,
                                                           8,9,10,11,12,13,14,15);
                s = wmma_bf16(qa[f], kb, s);
            }
            sc[kt] = s;
        }

        // ---- softmax over keys (row = v + 8*half lives across one 16-lane
        //      half-wave; reduce with xor shuffles inside that half) ----
        float inv[8];
        #pragma unroll
        for (int v = 0; v < 8; ++v) {
            float m = -1e30f;
            #pragma unroll
            for (int kt = 0; kt < KT; ++kt)
                m = fmaxf(m, sc[kt][v] * scale);   // padded cols are 0: safe upper bound
            #pragma unroll
            for (int off = 1; off < 16; off <<= 1)
                m = fmaxf(m, __shfl_xor(m, off, 32));
            float s = 0.f;
            #pragma unroll
            for (int kt = 0; kt < KT; ++kt) {
                const bool valid = (kt < 12) | (r < 5);   // col = kt*16 + r < 197
                float e = valid ? __expf(sc[kt][v] * scale - m) : 0.f;
                sc[kt][v] = e;
                s += e;
            }
            #pragma unroll
            for (int off = 1; off < 16; off <<= 1)
                s += __shfl_xor(s, off, 32);
            inv[v] = 1.f / s;
        }

        // ---- write normalized probabilities bf16 to LDS (A-layout source) --
        #pragma unroll
        for (int kt = 0; kt < KT; ++kt)
            #pragma unroll
            for (int v = 0; v < 8; ++v)
                sP[w][v + 8 * half][kt * 16 + r] = (__bf16)(sc[kt][v] * inv[v]);

        // ---- O = P(16 x 224) * V(224 x 64) ----
        v8f oacc[4] = {};
        #pragma unroll
        for (int kk = 0; kk < KP2 / 32; ++kk) {
            v16bf pa = frag_ld(&sP[w][0][kk * 32], SVL, lane);
            #pragma unroll
            for (int j = 0; j < 4; ++j) {
                v16bf vb = frag_ld(&sVT[j * 16][kk * 32], SVL, lane);
                oacc[j] = wmma_bf16(pa, vb, oacc[j]);
            }
        }

        // ---- store O tile ----
        const int crow = half << 3;
        #pragma unroll
        for (int j = 0; j < 4; ++j)
            #pragma unroll
            for (int v = 0; v < 8; ++v) {
                const int qr = n0 + crow + v;
                if (qr < NTOK)
                    o[((size_t)(b * NTOK + qr)) * DMODEL + hd * DH + j * 16 + r]
                        = (__bf16)oacc[j][v];
            }
    }
}

// ---------------------------------------------------------------------------
// LayerNorm: one wave32 per row of 768; outputs bf16 (GEMM input) or fp32.
// ---------------------------------------------------------------------------
__global__ __launch_bounds__(256)
void k_ln(const float* __restrict__ x, const float* __restrict__ g,
          const float* __restrict__ bta, __bf16* __restrict__ y16,
          float* __restrict__ y32, int rows)
{
    const int w    = threadIdx.x >> 5;
    const int lane = threadIdx.x & 31;
    const int row  = blockIdx.x * 8 + w;
    if (row >= rows) return;

    const float* xr = x + (size_t)row * DMODEL;
    float vals[DMODEL / 32];
    float s = 0.f;
    #pragma unroll
    for (int i = 0; i < DMODEL / 32; ++i) {
        vals[i] = xr[lane + 32 * i];
        s += vals[i];
    }
    #pragma unroll
    for (int off = 16; off > 0; off >>= 1) s += __shfl_xor(s, off, 32);
    const float mu = s * (1.f / DMODEL);

    float q = 0.f;
    #pragma unroll
    for (int i = 0; i < DMODEL / 32; ++i) {
        const float d = vals[i] - mu;
        q += d * d;
    }
    #pragma unroll
    for (int off = 16; off > 0; off >>= 1) q += __shfl_xor(q, off, 32);
    const float rstd = rsqrtf(q * (1.f / DMODEL) + 1e-6f);

    #pragma unroll
    for (int i = 0; i < DMODEL / 32; ++i) {
        const int c = lane + 32 * i;
        const float o = (vals[i] - mu) * rstd * g[c] + bta[c];
        if (y16) y16[(size_t)row * DMODEL + c] = (__bf16)o;
        if (y32) y32[(size_t)row * DMODEL + c] = o;
    }
}

// ---------------------------------------------------------------------------
// im2col for 16x16 non-overlapping patches: [B*196, 768] bf16, k = (c,p,q)
// ---------------------------------------------------------------------------
__global__ void k_im2col(const float* __restrict__ x, __bf16* __restrict__ pm)
{
    const long long idx = (long long)blockIdx.x * 256 + threadIdx.x;
    if (idx >= (long long)B_SZ * 196 * DMODEL) return;
    const int m  = (int)(idx / DMODEL);
    const int k  = (int)(idx % DMODEL);
    const int b  = m / 196, t2 = m % 196;
    const int gy = t2 / GRID_P, gx = t2 % GRID_P;
    const int c  = k >> 8, pr = (k >> 4) & 15, q = k & 15;
    const float v = x[(((size_t)b * 3 + c) * 224 + gy * 16 + pr) * 224 + gx * 16 + q];
    pm[idx] = (__bf16)v;
}

// ---------------------------------------------------------------------------
// Insert cls token row and add positional embedding.
// ---------------------------------------------------------------------------
__global__ void k_finish_embed(float* __restrict__ h, const float* __restrict__ cls,
                               const float* __restrict__ pos)
{
    const long long idx = (long long)blockIdx.x * 256 + threadIdx.x;
    if (idx >= (long long)MROWS * DMODEL) return;
    const int d  = (int)(idx % DMODEL);
    const int rn = (int)((idx / DMODEL) % NTOK);
    if (rn == 0) h[idx] = cls[d] + pos[d];
    else         h[idx] += pos[(size_t)rn * DMODEL + d];
}

// ---------------------------------------------------------------------------
// Host-side launch
// ---------------------------------------------------------------------------
extern "C" void kernel_launch(void* const* d_in, const int* in_sizes, int n_in,
                              void* d_out, int out_size, void* d_ws, size_t ws_size,
                              hipStream_t stream)
{
    (void)in_sizes; (void)n_in; (void)out_size; (void)ws_size;

    const float* x       = (const float*)d_in[0];
    const float* conv_w  = (const float*)d_in[1];
    const float* conv_b  = (const float*)d_in[2];
    const float* cls_tok = (const float*)d_in[3];
    const float* pos_emb = (const float*)d_in[4];
    const float* ln1_g   = (const float*)d_in[5];
    const float* ln1_b   = (const float*)d_in[6];
    const float* qkv_w   = (const float*)d_in[7];
    const float* qkv_b   = (const float*)d_in[8];
    const float* proj_w  = (const float*)d_in[9];
    const float* proj_b  = (const float*)d_in[10];
    const float* ln2_g   = (const float*)d_in[11];
    const float* ln2_b   = (const float*)d_in[12];
    const float* fc1_w   = (const float*)d_in[13];
    const float* fc1_b   = (const float*)d_in[14];
    const float* fc2_w   = (const float*)d_in[15];
    const float* fc2_b   = (const float*)d_in[16];
    const float* norm_g  = (const float*)d_in[17];
    const float* norm_b  = (const float*)d_in[18];

    auto au = [](size_t v) { return (v + 255) & ~(size_t)255; };
    char* ws = (char*)d_ws;
    float*  h       = (float*)ws;  ws += au((size_t)MROWS * DMODEL * 4);
    __bf16* ybuf    = (__bf16*)ws; ws += au((size_t)MROWS * DMODEL * 2);
    __bf16* qkvb    = (__bf16*)ws; ws += au((size_t)MROWS * 3 * DMODEL * 2);
    __bf16* obuf    = (__bf16*)ws; ws += au((size_t)MROWS * DMODEL * 2);
    __bf16* mbuf    = (__bf16*)ws; ws += au((size_t)MROWS * FF_DIM * 2);
    __bf16* patches = (__bf16*)ws; ws += au((size_t)B_SZ * 196 * DMODEL * 2);

    const int M = MROWS;                       // 6304
    const int gM = (M + BM - 1) / BM;          // 50

    // ---- patch embedding ----
    {
        const long long ne = (long long)B_SZ * 196 * DMODEL;
        k_im2col<<<(unsigned)((ne + 255) / 256), 256, 0, stream>>>(x, patches);
        k_gemm<<<dim3(196 * B_SZ / BM, DMODEL / BN), 256, 0, stream>>>(
            patches, conv_w, conv_b, nullptr, h, nullptr,
            196 * B_SZ, DMODEL, DMODEL, /*flags=*/2);
        const long long nh = (long long)MROWS * DMODEL;
        k_finish_embed<<<(unsigned)((nh + 255) / 256), 256, 0, stream>>>(h, cls_tok, pos_emb);
    }

    // ---- transformer layers ----
    for (int l = 0; l < DEPTH; ++l) {
        k_ln<<<(M + 7) / 8, 256, 0, stream>>>(
            h, ln1_g + l * DMODEL, ln1_b + l * DMODEL, ybuf, nullptr, M);

        k_gemm<<<dim3(gM, 3 * DMODEL / BN), 256, 0, stream>>>(
            ybuf, qkv_w + (size_t)l * 3 * DMODEL * DMODEL, qkv_b + l * 3 * DMODEL,
            nullptr, nullptr, qkvb, M, 3 * DMODEL, DMODEL, 0);

        k_attn<<<B_SZ * HEADS, 128, 0, stream>>>(qkvb, obuf, 0.125f);

        k_gemm<<<dim3(gM, DMODEL / BN), 256, 0, stream>>>(
            obuf, proj_w + (size_t)l * DMODEL * DMODEL, proj_b + l * DMODEL,
            h, h, nullptr, M, DMODEL, DMODEL, 0);

        k_ln<<<(M + 7) / 8, 256, 0, stream>>>(
            h, ln2_g + l * DMODEL, ln2_b + l * DMODEL, ybuf, nullptr, M);

        k_gemm<<<dim3(gM, FF_DIM / BN), 256, 0, stream>>>(
            ybuf, fc1_w + (size_t)l * FF_DIM * DMODEL, fc1_b + l * FF_DIM,
            nullptr, nullptr, mbuf, M, FF_DIM, DMODEL, /*flags=*/1);

        k_gemm<<<dim3(gM, DMODEL / BN), 256, 0, stream>>>(
            mbuf, fc2_w + (size_t)l * DMODEL * FF_DIM, fc2_b + l * DMODEL,
            h, h, nullptr, M, DMODEL, FF_DIM, 0);
    }

    // ---- final layernorm -> d_out (fp32) ----
    k_ln<<<(M + 7) / 8, 256, 0, stream>>>(h, norm_g, norm_b, nullptr,
                                          (float*)d_out, M);
}
